// CustomAttentionFFN_32315333935767
// MI455X (gfx1250) — matrix operand order repaired and checked by reading, not compiled, not run on hardware
//
#include <hip/hip_runtime.h>

typedef unsigned short u16;
typedef unsigned int   u32;
typedef __bf16 bf16;
typedef bf16  bf16x16 __attribute__((ext_vector_type(16)));
typedef float f32x8   __attribute__((ext_vector_type(8)));

#define B_     2
#define T_     1024
#define D_     1024
#define SLOTS_ 4
#define S_     (T_ * SLOTS_)   // 4096
#define MEXP_  (B_ * T_)       // 2048
#define MQKV_  (B_ * S_)       // 8192

// ---------- helpers ----------
__device__ __forceinline__ u16 f2bf(float f) {
  u32 u = __builtin_bit_cast(u32, f);
  u32 r = 0x7fffu + ((u >> 16) & 1u);       // round-to-nearest-even
  return (u16)((u + r) >> 16);
}
__device__ __forceinline__ float bf2f(u16 h) {
  u32 u = ((u32)h) << 16;
  return __builtin_bit_cast(float, u);
}

union BV { uint4 u[2]; bf16x16 v; };
union F8 { f32x8 v; float4 q[2]; };

// 16 consecutive bf16 (B-operand per-lane slice: K = hl?16..31:0..15 contiguous)
__device__ __forceinline__ bf16x16 load_b_op(const u16* p) {
  BV t; t.u[0] = *(const uint4*)(p); t.u[1] = *(const uint4*)(p + 8); return t.v;
}
// A-operand per-lane slice: 16 consecutive bf16 (K = hl?16..31:0..15 contiguous)
__device__ __forceinline__ bf16x16 load_a_op(const u16* p) {
  BV t; t.u[0] = *(const uint4*)(p); t.u[1] = *(const uint4*)(p + 8); return t.v;
}

__device__ __forceinline__ f32x8 wmma_bf16(bf16x16 a, bf16x16 b, f32x8 c) {
  return __builtin_amdgcn_wmma_f32_16x16x32_bf16(false, a, false, b, (short)0, c,
                                                 false, false);
}
__device__ __forceinline__ f32x8 zero8() {
  f32x8 z = {0.f, 0.f, 0.f, 0.f, 0.f, 0.f, 0.f, 0.f}; return z;
}
__device__ __forceinline__ float gelu_exact(float x) {
  return 0.5f * x * (1.0f + erff(x * 0.70710678118654752f));
}

// ---------- conversion kernels ----------
__global__ void k_cvt_bf16(const float* __restrict__ in, u16* __restrict__ out, int n4) {
  int i = blockIdx.x * blockDim.x + threadIdx.x;
  if (i < n4) {
    float4 f = ((const float4*)in)[i];
    u32 lo = (u32)f2bf(f.x) | ((u32)f2bf(f.y) << 16);
    u32 hi = (u32)f2bf(f.z) | ((u32)f2bf(f.w) << 16);
    ((uint2*)out)[i] = make_uint2(lo, hi);
  }
}

// in: f32 [R][C]  ->  out: bf16 [C][R]
__global__ void k_cvt_transpose(const float* __restrict__ in, u16* __restrict__ out,
                                int R, int C) {
  __shared__ float tile[32][33];
  int c0 = blockIdx.x * 32, r0 = blockIdx.y * 32;
  int tx = threadIdx.x, ty = threadIdx.y;   // block (32,8)
#pragma unroll
  for (int i = 0; i < 32; i += 8)
    tile[ty + i][tx] = in[(size_t)(r0 + ty + i) * C + (c0 + tx)];
  __syncthreads();
#pragma unroll
  for (int i = 0; i < 32; i += 8)
    out[(size_t)(c0 + ty + i) * R + (r0 + tx)] = f2bf(tile[tx][ty + i]);
}

// bf16 [R][C] -> bf16 [C][R], per-batch via blockIdx.z
__global__ void k_transpose_bf16(const u16* __restrict__ in, u16* __restrict__ out,
                                 int R, int C) {
  __shared__ u16 tile[32][34];
  size_t boff = (size_t)blockIdx.z * R * C;
  in += boff; out += boff;
  int c0 = blockIdx.x * 32, r0 = blockIdx.y * 32;
  int tx = threadIdx.x, ty = threadIdx.y;   // block (32,8)
#pragma unroll
  for (int i = 0; i < 32; i += 8)
    tile[ty + i][tx] = in[(size_t)(r0 + ty + i) * C + (c0 + tx)];
  __syncthreads();
#pragma unroll
  for (int i = 0; i < 32; i += 8)
    out[(size_t)(c0 + ty + i) * R + (r0 + tx)] = tile[tx][ty + i];
}

// ---------- WMMA GEMM: C = act(A[M,K] * Bt[N,K]^T + bias) ----------
// EPI 0: GELU + slot remap, bf16 out.  EPI 1: bf16 out.  EPI 2: f32 out.
template <int EPI>
__global__ __launch_bounds__(256)
void k_gemm_bf16(const u16* __restrict__ A, const u16* __restrict__ Bt,
                 const float* __restrict__ bias, void* __restrict__ outv,
                 int M, int N, int K) {
  const int lane   = threadIdx.x & 31;
  const int wid    = threadIdx.x >> 5;
  const int lane15 = lane & 15;
  const int hl     = lane >> 4;
  const int rowBase = blockIdx.y * 128 + (wid >> 1) * 32;  // 4 wave rows
  const int colBase = blockIdx.x * 128 + (wid & 1) * 64;   // 2 wave cols

  f32x8 acc[2][4];
#pragma unroll
  for (int i = 0; i < 2; i++)
#pragma unroll
    for (int j = 0; j < 4; j++) acc[i][j] = zero8();

  const int aoff = hl * 16;
  const int boff = hl * 16;
  for (int k0 = 0; k0 < K; k0 += 32) {
    bf16x16 a0 = load_a_op(A + (size_t)(rowBase + lane15) * K + k0 + aoff);
    bf16x16 a1 = load_a_op(A + (size_t)(rowBase + 16 + lane15) * K + k0 + aoff);
#pragma unroll
    for (int tn = 0; tn < 4; tn++) {
      bf16x16 b = load_b_op(Bt + (size_t)(colBase + tn * 16 + lane15) * K + k0 + boff);
      acc[0][tn] = wmma_bf16(a0, b, acc[0][tn]);
      acc[1][tn] = wmma_bf16(a1, b, acc[1][tn]);
    }
  }

#pragma unroll
  for (int tn = 0; tn < 4; tn++) {
    const int col = colBase + tn * 16 + lane15;
    const float bv = bias[col];
#pragma unroll
    for (int tm = 0; tm < 2; tm++) {
#pragma unroll
      for (int r = 0; r < 8; r++) {
        const int row = rowBase + tm * 16 + r + 8 * hl;
        float v = acc[tm][tn][r] + bv;
        if constexpr (EPI == 0) {           // expand: GELU + fold slots into S
          v = gelu_exact(v);
          int b = row >> 10, t = row & 1023;       // row in [0, B*T)
          int slot = col >> 10, d = col & 1023;    // col in [0, SLOTS*D)
          size_t idx = (((size_t)(b * S_ + t * SLOTS_ + slot)) << 10) + d;
          ((u16*)outv)[idx] = f2bf(v);
        } else if constexpr (EPI == 1) {
          ((u16*)outv)[(size_t)row * N + col] = f2bf(v);
        } else {
          ((float*)outv)[(size_t)row * N + col] = v;
        }
      }
    }
  }
}

// ---------- Flash attention ----------
// 16 q-rows per workgroup, 16 waves (512 threads). Wave w owns a 64-wide D slice
// of the QK^T contraction and 64 output columns. Wave 0 performs the softmax.
// Row stats live in LDS; wave-0 reduction unroll is capped to keep its register
// peak low so the Q operands stay resident (no scratch spills).
__global__ __launch_bounds__(512)
void k_flash_attn(const u16* __restrict__ q, const u16* __restrict__ k,
                  const u16* __restrict__ vt, u16* __restrict__ mixed) {
  __shared__ __align__(16) float sS[16][2][256];  // 32 KB partial score tiles
  __shared__ __align__(16) u16   sP[16][40];      // 16x32 bf16 probabilities (+pad)
  __shared__ float sCorr[16];                     // per-row rescale / final 1/l
  __shared__ float sM[16], sL[16];                // online softmax row stats

  const int lane   = threadIdx.x & 31;
  const int w      = threadIdx.x >> 5;
  const int lane15 = lane & 15;
  const int hl     = lane >> 4;
  const int s0     = blockIdx.x * 16;
  const int b      = blockIdx.y;
  const u16* qb = q  + (size_t)b * S_ * D_;
  const u16* kb = k  + (size_t)b * S_ * D_;
  const u16* vb = vt + (size_t)b * D_ * S_;   // [D][S]
  const int colBase = w * 64;

  if (w == 0 && lane < 16) { sM[lane] = -1e30f; sL[lane] = 0.f; }
  __syncthreads();

  // Q A-operands for this wave's 64-wide D slice
  bf16x16 qa[2];
#pragma unroll
  for (int kk = 0; kk < 2; kk++)
    qa[kk] = load_a_op(qb + (size_t)(s0 + lane15) * D_ + w * 64 + kk * 32 + hl * 16);

  f32x8 o[4];
#pragma unroll
  for (int j = 0; j < 4; j++) o[j] = zero8();

  // per-lane global bases (row fixed by lane)
  const u16* kRow0 = kb + (size_t)lane15 * D_ + w * 64 + hl * 16;        // + js*D_ + kk*32
  const u16* kRow1 = kb + (size_t)(16 + lane15) * D_ + w * 64 + hl * 16;
  const u16* vRow  = vb + (size_t)(colBase + lane15) * S_ + hl * 16;     // + j*16*S_ + js

  for (int js = 0; js < S_; js += 32) {
    // prefetch next K/V tiles (global_prefetch_b8)
    if (js + 32 < S_) {
      __builtin_prefetch(kRow0 + (size_t)(js + 32) * D_, 0, 3);
      __builtin_prefetch(vRow + (size_t)js + 32, 0, 3);
    }

    // ---- partial scores for this wave's D slice: S[16 x 32] ----
    // batch all 4 operand loads (one clause), then 4 WMMAs with staggered waits
    f32x8 sa[2] = {zero8(), zero8()};
    {
      bf16x16 bb[4];
      bb[0] = load_b_op(kRow0 + (size_t)js * D_);
      bb[1] = load_b_op(kRow1 + (size_t)js * D_);
      bb[2] = load_b_op(kRow0 + (size_t)js * D_ + 32);
      bb[3] = load_b_op(kRow1 + (size_t)js * D_ + 32);
      sa[0] = wmma_bf16(qa[0], bb[0], sa[0]);
      sa[1] = wmma_bf16(qa[0], bb[1], sa[1]);
      sa[0] = wmma_bf16(qa[1], bb[2], sa[0]);
      sa[1] = wmma_bf16(qa[1], bb[3], sa[1]);
    }
#pragma unroll
    for (int tn = 0; tn < 2; tn++) {
      F8 t; t.v = sa[tn];
      *(float4*)&sS[w][tn][lane * 8]     = t.q[0];
      *(float4*)&sS[w][tn][lane * 8 + 4] = t.q[1];
    }
    __syncthreads();

    // ---- wave 0: reduce partials + online softmax; publish P and corr ----
    if (w == 0) {
      f32x8 v0a = zero8(), v1a = zero8();
#pragma unroll 2      // cap concurrent LDS loads: keeps this branch's VGPR peak low
      for (int ww = 0; ww < 16; ww++) {
        F8 t0, t1;
        t0.q[0] = *(const float4*)&sS[ww][0][lane * 8];
        t0.q[1] = *(const float4*)&sS[ww][0][lane * 8 + 4];
        t1.q[0] = *(const float4*)&sS[ww][1][lane * 8];
        t1.q[1] = *(const float4*)&sS[ww][1][lane * 8 + 4];
        v0a = v0a + t0.v;
        v1a = v1a + t1.v;
      }
#pragma unroll
      for (int r = 0; r < 8; r++) {
        float v0 = v0a[r] * 0.03125f;          // 1/sqrt(1024)
        float v1 = v1a[r] * 0.03125f;
        float rm = fmaxf(v0, v1);
#pragma unroll
        for (int m = 1; m < 16; m <<= 1) rm = fmaxf(rm, __shfl_xor(rm, m, 32));
        int row = r + 8 * hl;
        float m_old = sM[row];                 // uniform per lane-half
        float l_old = sL[row];
        float mn = fmaxf(m_old, rm);
        float c  = __expf(m_old - mn);
        float e0 = __expf(v0 - mn);
        float e1 = __expf(v1 - mn);
        sP[row][lane15]      = f2bf(e0);
        sP[row][16 + lane15] = f2bf(e1);
        float ps = e0 + e1;
#pragma unroll
        for (int m = 1; m < 16; m <<= 1) ps += __shfl_xor(ps, m, 32);
        if (lane15 == 0) {
          sCorr[row] = c;
          sM[row] = mn;
          sL[row] = l_old * c + ps;
        }
      }
    }
    __syncthreads();

    // ---- all waves: rescale O, then O[:, 64 cols] += P @ V ----
    f32x8 corrv;
#pragma unroll
    for (int r = 0; r < 8; r++) corrv[r] = sCorr[r + 8 * hl];
#pragma unroll
    for (int j = 0; j < 4; j++) o[j] = o[j] * corrv;

    bf16x16 pa = load_a_op(&sP[lane15][hl * 16]);
    {
      bf16x16 bb[4];
#pragma unroll
      for (int j = 0; j < 4; j++)
        bb[j] = load_b_op(vRow + (size_t)js + (size_t)j * 16 * S_);
#pragma unroll
      for (int j = 0; j < 4; j++)
        o[j] = wmma_bf16(pa, bb[j], o[j]);
    }
    __syncthreads();   // sS/sP reused next iteration
  }

  // ---- wave 0 publishes 1/l; all waves normalize and write ----
  if (w == 0 && lane < 16) sCorr[lane] = 1.0f / sL[lane];
  __syncthreads();
  f32x8 invv;
#pragma unroll
  for (int r = 0; r < 8; r++) invv[r] = sCorr[r + 8 * hl];
#pragma unroll
  for (int j = 0; j < 4; j++) {
    f32x8 ov = o[j] * invv;
#pragma unroll
    for (int r = 0; r < 8; r++) {
      int row = s0 + r + 8 * hl;
      int d = colBase + j * 16 + lane15;
      mixed[((size_t)b * S_ + row) * D_ + d] = f2bf(ov[r]);
    }
  }
}

// ---------- slot mean: [B, S, D] -> [B, T, D] ----------
__global__ void k_slot_avg(const u16* __restrict__ mix, u16* __restrict__ avg) {
  int i = blockIdx.x * blockDim.x + threadIdx.x;   // over B*T*D
  if (i < MEXP_ * D_) {
    int d  = i & 1023;
    int tp = i >> 10;               // b*T + t
    int b = tp >> 10, t = tp & 1023;
    size_t base = ((size_t)(b * S_ + t * 4)) * D_ + d;
    float s = bf2f(mix[base]) + bf2f(mix[base + D_]) +
              bf2f(mix[base + 2 * D_]) + bf2f(mix[base + 3 * D_]);
    avg[i] = f2bf(0.25f * s);
  }
}

// ---------- launch ----------
extern "C" void kernel_launch(void* const* d_in, const int* in_sizes, int n_in,
                              void* d_out, int out_size, void* d_ws, size_t ws_size,
                              hipStream_t stream) {
  (void)in_sizes; (void)n_in; (void)out_size; (void)ws_size;
  const float* tokens = (const float*)d_in[0];
  const float* We = (const float*)d_in[1];
  const float* be = (const float*)d_in[2];
  const float* Wq = (const float*)d_in[3];
  const float* bq = (const float*)d_in[4];
  const float* Wk = (const float*)d_in[5];
  const float* bk = (const float*)d_in[6];
  const float* Wv = (const float*)d_in[7];
  const float* bv = (const float*)d_in[8];
  const float* Wo = (const float*)d_in[9];
  const float* bo = (const float*)d_in[10];
  float* out = (float*)d_out;

  char* ws = (char*)d_ws;
  const size_t MB = 1024ull * 1024ull;
  u16* tok_bf = (u16*)(ws + 0);        //  4MB (later reused as avg)
  u16* wet    = (u16*)(ws + 4 * MB);   //  8MB  We^T  [4096,1024]
  u16* wqt    = (u16*)(ws + 12 * MB);  //  2MB
  u16* wkt    = (u16*)(ws + 14 * MB);  //  2MB
  u16* wvt    = (u16*)(ws + 16 * MB);  //  2MB
  u16* wot    = (u16*)(ws + 18 * MB);  //  2MB
  u16* expb   = (u16*)(ws + 20 * MB);  // 16MB expanded (later reused as vt)
  u16* qbuf   = (u16*)(ws + 36 * MB);  // 16MB
  u16* kbuf   = (u16*)(ws + 52 * MB);  // 16MB
  u16* vbuf   = (u16*)(ws + 68 * MB);  // 16MB (later reused as mixed)
  u16* vtbuf  = expb;                  // alias: expanded dead after QKV GEMMs
  u16* mixbuf = vbuf;                  // alias: v dead after transpose
  u16* avgbuf = tok_bf;                // alias: tokens_bf dead after expand

  // 1) fp32 -> bf16 conversions (tokens; weights converted + transposed)
  {
    int n4 = (B_ * T_ * D_) / 4;
    k_cvt_bf16<<<dim3((n4 + 255) / 256), dim3(256), 0, stream>>>(tokens, tok_bf, n4);
  }
  k_cvt_transpose<<<dim3(4096 / 32, 1024 / 32), dim3(32, 8), 0, stream>>>(We, wet, 1024, 4096);
  k_cvt_transpose<<<dim3(32, 32), dim3(32, 8), 0, stream>>>(Wq, wqt, 1024, 1024);
  k_cvt_transpose<<<dim3(32, 32), dim3(32, 8), 0, stream>>>(Wk, wkt, 1024, 1024);
  k_cvt_transpose<<<dim3(32, 32), dim3(32, 8), 0, stream>>>(Wv, wvt, 1024, 1024);
  k_cvt_transpose<<<dim3(32, 32), dim3(32, 8), 0, stream>>>(Wo, wot, 1024, 1024);

  // 2) expand GEMM + GELU + slot fold:  [2048,1024] x [1024,4096] -> [B,S,D] bf16
  k_gemm_bf16<0><<<dim3(4096 / 128, MEXP_ / 128), 256, 0, stream>>>(
      tok_bf, wet, be, expb, MEXP_, SLOTS_ * D_, D_);

  // 3) Q, K, V GEMMs: [8192,1024] x [1024,1024]
  k_gemm_bf16<1><<<dim3(1024 / 128, MQKV_ / 128), 256, 0, stream>>>(
      expb, wqt, bq, qbuf, MQKV_, D_, D_);
  k_gemm_bf16<1><<<dim3(1024 / 128, MQKV_ / 128), 256, 0, stream>>>(
      expb, wkt, bk, kbuf, MQKV_, D_, D_);
  k_gemm_bf16<1><<<dim3(1024 / 128, MQKV_ / 128), 256, 0, stream>>>(
      expb, wvt, bv, vbuf, MQKV_, D_, D_);

  // 4) V -> V^T per batch ([S,D] -> [D,S]) for contiguous PV B-operands
  k_transpose_bf16<<<dim3(D_ / 32, S_ / 32, B_), dim3(32, 8), 0, stream>>>(
      vbuf, vtbuf, S_, D_);

  // 5) attention (flash-style, exact softmax), 512 threads / 16 waves per WG
  k_flash_attn<<<dim3(S_ / 16, B_), 512, 0, stream>>>(qbuf, kbuf, vtbuf, mixbuf);

  // 6) slot mean, then output projection in fp32 straight to d_out
  k_slot_avg<<<dim3((MEXP_ * D_) / 256), 256, 0, stream>>>(mixbuf, avgbuf);
  k_gemm_bf16<2><<<dim3(1024 / 128, MEXP_ / 128), 256, 0, stream>>>(
      avgbuf, wot, bo, out, MEXP_, D_, D_);
}
